// EFNGlobal_76441827934549
// MI455X (gfx1250) — compile-verified, matrix operand order checked by reading, not run on hardware
//
#include <hip/hip_runtime.h>
#include <hip/hip_bf16.h>

typedef __bf16        v16bf  __attribute__((ext_vector_type(16)));
typedef float         v8f    __attribute__((ext_vector_type(8)));
typedef unsigned int  uint4v __attribute__((ext_vector_type(4)));

#define D_FEAT   256
#define D_IN     512    // 2*D_FEAT
#define D_HID    512
#define NG       16
#define NSLICE   128
#define TPB      512    // 16 waves
#define TPB_AUX  256
#define E_WG     48     // edges per workgroup (3 M-tiles of 16)
#define NMT      3      // M-tiles per workgroup
#define LDSW     520    // 512 + 8 ushort pad -> bank-conflict-free row stride

__device__ __forceinline__ unsigned short f2bf(float f) {
    unsigned int u = __float_as_uint(f);
    u += 0x7FFFu + ((u >> 16) & 1u);      // round-to-nearest-even
    return (unsigned short)(u >> 16);
}

union FragU { v16bf v; uint4v u[2]; };

__device__ __forceinline__ v16bf load_frag(const unsigned short* p0,
                                           const unsigned short* p1) {
    FragU f;
    f.u[0] = *(const uint4v*)p0;   // 8 bf16 (K..K+7)
    f.u[1] = *(const uint4v*)p1;   // 8 bf16 (K+16..K+23)
    return f.v;
}

// ---- prologue: W1 (512x512) / W2 (512x256) f32 row-major -> bf16 transposed [N][K]
__global__ void efn_convert(const float* __restrict__ W1, const float* __restrict__ W2,
                            unsigned short* __restrict__ W1t, unsigned short* __restrict__ W2t) {
    int idx = blockIdx.x * blockDim.x + threadIdx.x;
    if (idx < D_HID * D_IN) {
        int n = idx / D_IN, k = idx % D_IN;
        W1t[n * D_IN + k] = f2bf(W1[k * D_HID + n]);
    } else {
        int j = idx - D_HID * D_IN;
        if (j < D_FEAT * D_HID) {
            int n = j / D_HID, k = j % D_HID;
            W2t[n * D_HID + k] = f2bf(W2[k * D_FEAT + n]);
        }
    }
}

__global__ void efn_zero(float* __restrict__ part, int n) {
    int i = blockIdx.x * blockDim.x + threadIdx.x;
    if (i < n) part[i] = 0.0f;
}

__global__ __launch_bounds__(TPB, 1)
void efn_main(const float* __restrict__ x, const float* __restrict__ p,
              const int* __restrict__ ei, const int* __restrict__ batch,
              const float* __restrict__ b1, const float* __restrict__ b2,
              const unsigned short* __restrict__ W1t, const unsigned short* __restrict__ W2t,
              float* __restrict__ part, int nE) {
    // single tile buffer: bf16 m [48][512] for layer 1, then aliased to h for layer 2
    __shared__ unsigned short tileAB[E_WG * LDSW];
    __shared__ float escale[E_WG];
    __shared__ int   egraph[E_WG];
    __shared__ int   sIdx[E_WG];
    __shared__ int   dIdx[E_WG];

    const int tid = threadIdx.x;
    const int e0  = blockIdx.x * E_WG;

    if (tid < E_WG) {
        int e = e0 + tid;
        int s = 0, d = 0, g = 0; float sc = 0.0f;
        if (e < nE) {
            s  = ei[e];            // edge_index[0][e] = src
            d  = ei[nE + e];       // edge_index[1][e] = dst
            sc = p[d * 4 + 0];     // p[:,0]
            g  = batch[d];
        }
        sIdx[tid] = s; dIdx[tid] = d; escale[tid] = sc; egraph[tid] = g;
    }
    __syncthreads();

    // ---- build m = [x_d, x_s - x_d] in bf16; strided float4 sweep (coalesced)
    {
        const int NQ = E_WG * (D_IN / 4);           // 48*128 = 6144 quads
        #pragma unroll
        for (int q = tid; q < NQ; q += TPB) {
            const int el = q >> 7;                  // edge
            const int qc = q & 127;                 // quad within row
            const int col = qc * 4;
            const int s = sIdx[el], d = dIdx[el];
            unsigned short* dst = tileAB + el * LDSW + col;
            if (col < D_FEAT) {
                float4 v = *(const float4*)(x + (size_t)d * D_FEAT + col);
                dst[0] = f2bf(v.x); dst[1] = f2bf(v.y);
                dst[2] = f2bf(v.z); dst[3] = f2bf(v.w);
            } else {
                float4 a = *(const float4*)(x + (size_t)s * D_FEAT + (col - D_FEAT));
                float4 b = *(const float4*)(x + (size_t)d * D_FEAT + (col - D_FEAT));
                dst[0] = f2bf(a.x - b.x); dst[1] = f2bf(a.y - b.y);
                dst[2] = f2bf(a.z - b.z); dst[3] = f2bf(a.w - b.w);
            }
        }
    }
    __syncthreads();

    const int lane = tid & 31;
    const int wv   = tid >> 5;          // wave 0..15
    const int l    = lane & 15;         // A: row / B: col / C,D: N
    const int off  = (lane >> 4) * 8;   // K half-offset per ISA 16-bit A/B layout
    const int rowb = (lane >> 4) * 8;   // C/D: M = vgpr_idx + rowb

    // ---- Layer 1: h = relu(m @ W1 + b1)
    // wave owns N-tiles wv*32 + {0,16}; 3 M-tiles reuse each B frag
    v8f acc[NMT][2];
    #pragma unroll
    for (int t = 0; t < 2; ++t) {
        float bv = b1[wv * 32 + t * 16 + l];
        #pragma unroll
        for (int mt = 0; mt < NMT; ++mt)
            #pragma unroll
            for (int r = 0; r < 8; ++r) acc[mt][t][r] = bv;
    }
    for (int k0 = 0; k0 < D_IN; k0 += 32) {
        v16bf a[NMT];
        #pragma unroll
        for (int mt = 0; mt < NMT; ++mt)
            a[mt] = load_frag(tileAB + (mt * 16 + l) * LDSW + k0 + off,
                              tileAB + (mt * 16 + l) * LDSW + k0 + 16 + off);
        #pragma unroll
        for (int t = 0; t < 2; ++t) {
            int n = wv * 32 + t * 16 + l;
            v16bf b = load_frag(W1t + n * D_IN + k0 + off,
                                W1t + n * D_IN + k0 + 16 + off);
            #pragma unroll
            for (int mt = 0; mt < NMT; ++mt)
                acc[mt][t] = __builtin_amdgcn_wmma_f32_16x16x32_bf16(
                                 false, a[mt], false, b, (short)0, acc[mt][t], false, false);
        }
    }
    __syncthreads();   // everyone done reading m before h overwrites the buffer

    #pragma unroll
    for (int mt = 0; mt < NMT; ++mt) {
        #pragma unroll
        for (int t = 0; t < 2; ++t) {
            int col = wv * 32 + t * 16 + l;
            #pragma unroll
            for (int r = 0; r < 8; ++r) {
                float hv = acc[mt][t][r];
                hv = hv > 0.0f ? hv : 0.0f;                       // relu
                tileAB[(mt * 16 + rowb + r) * LDSW + col] = f2bf(hv);
            }
        }
    }
    __syncthreads();

    // ---- Layer 2: msg = h @ W2 + b2; wave owns N-tile wv*16, 3 M-tiles
    v8f acc2[NMT];
    {
        float bv = b2[wv * 16 + l];
        #pragma unroll
        for (int mt = 0; mt < NMT; ++mt)
            #pragma unroll
            for (int r = 0; r < 8; ++r) acc2[mt][r] = bv;
    }
    for (int k0 = 0; k0 < D_HID; k0 += 32) {
        v16bf a[NMT];
        #pragma unroll
        for (int mt = 0; mt < NMT; ++mt)
            a[mt] = load_frag(tileAB + (mt * 16 + l) * LDSW + k0 + off,
                              tileAB + (mt * 16 + l) * LDSW + k0 + 16 + off);
        int n = wv * 16 + l;
        v16bf b = load_frag(W2t + n * D_HID + k0 + off,
                            W2t + n * D_HID + k0 + 16 + off);
        #pragma unroll
        for (int mt = 0; mt < NMT; ++mt)
            acc2[mt] = __builtin_amdgcn_wmma_f32_16x16x32_bf16(
                           false, a[mt], false, b, (short)0, acc2[mt], false, false);
    }

    // ---- epilogue: out[batch[dst]] += p[dst,0] * msg, staged into NSLICE partials
    float* pslice = part + (size_t)(blockIdx.x & (NSLICE - 1)) * (NG * D_FEAT);
    const int col = wv * 16 + l;
    #pragma unroll
    for (int mt = 0; mt < NMT; ++mt) {
        #pragma unroll
        for (int r = 0; r < 8; ++r) {
            int er = mt * 16 + rowb + r;                 // edge row in tile
            float val = acc2[mt][r] * escale[er];        // 0 for padded edges
            atomicAdd(&pslice[egraph[er] * D_FEAT + col], val);
        }
    }
}

__global__ void efn_reduce(const float* __restrict__ part, float* __restrict__ out) {
    int i = blockIdx.x * blockDim.x + threadIdx.x;       // 0..4095
    if (i < NG * D_FEAT) {
        float s = 0.0f;
        for (int sl = 0; sl < NSLICE; ++sl) s += part[(size_t)sl * (NG * D_FEAT) + i];
        out[i] = s;
    }
}

extern "C" void kernel_launch(void* const* d_in, const int* in_sizes, int n_in,
                              void* d_out, int out_size, void* d_ws, size_t ws_size,
                              hipStream_t stream) {
    (void)n_in; (void)out_size; (void)ws_size;
    const float* x    = (const float*)d_in[0];
    const float* p    = (const float*)d_in[1];
    const int*   ei   = (const int*)d_in[2];
    const int*   batch= (const int*)d_in[3];
    const float* W1   = (const float*)d_in[4];
    const float* b1   = (const float*)d_in[5];
    const float* W2   = (const float*)d_in[6];
    const float* b2   = (const float*)d_in[7];
    float*       out  = (float*)d_out;
    const int nE = in_sizes[2] / 2;

    // workspace layout: bf16 W1t (512 KB) | bf16 W2t (256 KB) | slice partials (2 MB)
    unsigned short* W1t = (unsigned short*)d_ws;
    unsigned short* W2t = W1t + D_HID * D_IN;
    float* part = (float*)((char*)d_ws +
                  (size_t)(D_HID * D_IN + D_FEAT * D_HID) * sizeof(unsigned short));

    const int nPart = NSLICE * NG * D_FEAT;
    efn_zero<<<(nPart + TPB_AUX - 1) / TPB_AUX, TPB_AUX, 0, stream>>>(part, nPart);

    const int nConv = D_HID * D_IN + D_FEAT * D_HID;
    efn_convert<<<(nConv + TPB_AUX - 1) / TPB_AUX, TPB_AUX, 0, stream>>>(W1, W2, W1t, W2t);

    const int nBlocks = (nE + E_WG - 1) / E_WG;
    efn_main<<<nBlocks, TPB, 0, stream>>>(x, p, ei, batch, b1, b2, W1t, W2t, part, nE);

    efn_reduce<<<(NG * D_FEAT + TPB_AUX - 1) / TPB_AUX, TPB_AUX, 0, stream>>>(part, out);
}
